// GeoformerMultiHeadAttention_63539746177657
// MI455X (gfx1250) — matrix operand order repaired and checked by reading, not compiled, not run on hardware
//
#include <hip/hip_runtime.h>

#define DEV __device__ __forceinline__

typedef _Float16 f16x4  __attribute__((ext_vector_type(4)));
typedef _Float16 f16x8  __attribute__((ext_vector_type(8)));
typedef _Float16 f16x16 __attribute__((ext_vector_type(16)));
typedef float    f32x8  __attribute__((ext_vector_type(8)));

constexpr int Bc = 8, Nc = 128, Ec = 256;
constexpr int ATTN_SZ = Bc * Nc * Ec;                 // 262144 floats
constexpr float CUTV = 5.0f;

// ---------------------------------------------------------------- helpers
DEV float siluf(float x) { return x / (1.0f + __expf(-x)); }

DEV float cos_cutoff(float d) {
  // 0.5*(cos(d*pi/5)+1) * (d<5)
  return 0.5f * (__cosf(d * 0.62831853071795864769f) + 1.0f) * (d < CUTV ? 1.0f : 0.0f);
}

// A fragment (16x32 f16, M=rows of tile, K from kbase..kbase+31), tile row-major [16][256] in LDS.
// Lanes 0-15 row=lane: half j -> K=kbase+16*hi+j (lanes 16-31: K shifted by +16).
DEV f16x16 load_a(const _Float16* tile, int l15, int hi, int kbase) {
  return *(const f16x16*)(tile + l15 * Ec + kbase + 16 * hi);
}

// B fragment (32x16 f16) from transposed weight WT[o][k]: lane half j -> B[K=kbase+16*hi+j, N=ct*16+l15]
DEV f16x16 load_b(const _Float16* wT, int ct, int kbase, int l15, int hi) {
  return *(const f16x16*)(wT + (ct * 16 + l15) * Ec + kbase + 16 * hi);
}

DEV f32x8 wmma(f16x16 a, f16x16 b, f32x8 c) {
  return __builtin_amdgcn_wmma_f32_16x16x32_f16(false, a, false, b, (short)0, c, false, false);
}

// stage 16 rows x 256 f32 from global -> f16 row-major LDS tile (one wave)
DEV void stage16x256(const float* __restrict__ src, _Float16* dst, int lane) {
  for (int i = lane; i < 1024; i += 32) {            // 1024 float4's
    float4 f = ((const float4*)src)[i];
    f16x4 h = {(_Float16)f.x, (_Float16)f.y, (_Float16)f.z, (_Float16)f.w};
    ((f16x4*)dst)[i] = h;
  }
}

// ---------------------------------------------------------------- kernels
// WT[o*256+k] = (f16) W[k*256+o]
__global__ void prep_wT(const float* __restrict__ W, _Float16* __restrict__ WT) {
  int o = blockIdx.x, k = threadIdx.x;
  WT[o * Ec + k] = (_Float16)W[k * Ec + o];
}

// q = (x*cm)@Wq + bq (f32);  k = x@Wk + bk (f16);  v = x@Wv + bv (f16)
__global__ __launch_bounds__(256) void proj_qkv(
    const float* __restrict__ x, const unsigned char* __restrict__ cmask,
    const _Float16* __restrict__ WqT, const _Float16* __restrict__ WkT, const _Float16* __restrict__ WvT,
    const float* __restrict__ bq, const float* __restrict__ bk, const float* __restrict__ bv,
    float* __restrict__ q32, _Float16* __restrict__ kh, _Float16* __restrict__ vh) {
  extern __shared__ char smem[];
  float* cm_s = (float*)smem;                           // 16 f32
  _Float16* x_h  = (_Float16*)(smem + 64);              // 16x256
  _Float16* xc_h = x_h + 16 * Ec;                       // 16x256
  int tid = threadIdx.x, w = tid >> 5, lane = tid & 31, l15 = lane & 15, hi = lane >> 4;
  int r0 = blockIdx.x * 16;

  if (tid < 16) cm_s[tid] = cmask[r0 + tid] ? 1.0f : 0.0f;
  __syncthreads();
  for (int i = tid; i < 1024; i += 256) {
    int row = i >> 6, c4 = i & 63;
    float4 f = ((const float4*)(x + (size_t)(r0 + row) * Ec))[c4];
    float sc = cm_s[row];
    f16x4 a  = {(_Float16)f.x, (_Float16)f.y, (_Float16)f.z, (_Float16)f.w};
    f16x4 ac = {(_Float16)(f.x * sc), (_Float16)(f.y * sc), (_Float16)(f.z * sc), (_Float16)(f.w * sc)};
    ((f16x4*)(x_h  + row * Ec))[c4] = a;
    ((f16x4*)(xc_h + row * Ec))[c4] = ac;
  }
  __syncthreads();

  for (int mat = 0; mat < 3; ++mat) {
    const _Float16* wT  = (mat == 0) ? WqT : (mat == 1) ? WkT : WvT;
    const float*    bia = (mat == 0) ? bq  : (mat == 1) ? bk  : bv;
    const _Float16* At  = (mat == 0) ? xc_h : x_h;
    for (int cs = 0; cs < 2; ++cs) {
      int ct = w + 8 * cs;
      f32x8 acc = {};
#pragma unroll
      for (int kb = 0; kb < 8; ++kb)
        acc = wmma(load_a(At, l15, hi, kb * 32), load_b(wT, ct, kb * 32, l15, hi), acc);
      int e = ct * 16 + l15;
      float bb = bia[e];
#pragma unroll
      for (int r = 0; r < 8; ++r) {
        int gr = r0 + r + 8 * hi;
        float v = acc[r] + bb;
        if (mat == 0)       q32[(size_t)gr * Ec + e] = v;
        else if (mat == 1)  kh[(size_t)gr * Ec + e] = (_Float16)v;
        else                vh[(size_t)gr * Ec + e] = (_Float16)v;
      }
    }
  }
}

// Fused pass 1: one block per (b,n). wave w owns m-tile rows [w*16, w*16+16).
__global__ __launch_bounds__(256) void pass1(
    const float* __restrict__ q32, const _Float16* __restrict__ kh, const _Float16* __restrict__ vh,
    const float* __restrict__ edge_attr, const float* __restrict__ dist,
    const unsigned char* __restrict__ kpm, const unsigned char* __restrict__ cmask,
    const float* __restrict__ vec,
    const _Float16* __restrict__ WdkT, const float* __restrict__ bdk,
    const _Float16* __restrict__ WduT, const float* __restrict__ bdu,
    const float* __restrict__ Wdih,
    float* __restrict__ cws, float* __restrict__ wtb, float* __restrict__ out) {
  extern __shared__ char smem[];
  float* qrow   = (float*)smem;          // 256
  float* vecs   = qrow + 256;            // 128*3
  float* du_acc = vecs + 384;            // 3*256
  float* probs  = du_acc + 768;          // 128*8 (reused as attn accumulator later)
  float* red    = probs + 1024;          // 8 (use [0])
  _Float16* ea_h  = (_Float16*)(red + 8);      // 128*256 f16
  _Float16* apn_h = ea_h + 128 * Ec;           // 128*256 f16 (dk, then apn)

  int tid = threadIdx.x, w = tid >> 5, lane = tid & 31, l15 = lane & 15, hi = lane >> 4;
  int bn = blockIdx.x, b = bn >> 7, mbase = w * 16;
  float cmv = cmask[bn] ? 1.0f : 0.0f;

  // ---- init + stage edge_attr tile (f32 -> f16)
  qrow[tid] = q32[(size_t)bn * Ec + tid];
  for (int i = tid; i < 384; i += 256) vecs[i] = vec[(size_t)bn * 384 + i];
  for (int i = tid; i < 768; i += 256) du_acc[i] = 0.0f;
  if (tid == 0) red[0] = 0.0f;
  stage16x256(edge_attr + (size_t)bn * Nc * Ec + (size_t)mbase * Ec, ea_h + mbase * Ec, lane);
  __syncthreads();

  // ---- dk = silu(ea @ Wdk + bdk)   (f16 WMMA, f32 accum) -> apn_h buffer as f16
  {
    const _Float16* At = ea_h + mbase * Ec;
    _Float16* Dt = apn_h + mbase * Ec;
    for (int ct = 0; ct < 16; ++ct) {
      f32x8 acc = {};
#pragma unroll
      for (int kb = 0; kb < 8; ++kb)
        acc = wmma(load_a(At, l15, hi, kb * 32), load_b(WdkT, ct, kb * 32, l15, hi), acc);
      int e = ct * 16 + l15;
      float bb = bdk[e];
#pragma unroll
      for (int r = 0; r < 8; ++r)
        Dt[(r + 8 * hi) * Ec + e] = (_Float16)siluf(acc[r] + bb);
    }
  }
  __syncthreads();

  // ---- aw[h][m] = sum_d q[e]*k[m][e]*dk[m][e];  probs = silu(aw*cm, masked)*cutoff(dist)
  for (int i = 0; i < 4; ++i) {
    int p = lane + 32 * i;
    int h = p >> 4, ml = p & 15, m = mbase + ml;
    const f16x8* dk8 = (const f16x8*)(apn_h + m * Ec + h * 32);
    const f16x8* k8  = (const f16x8*)(kh + (size_t)(b * Nc + m) * Ec + h * 32);
    const float* qp  = qrow + h * 32;
    float s = 0.0f;
#pragma unroll
    for (int c = 0; c < 4; ++c) {
      f16x8 kv = k8[c], dv = dk8[c];
#pragma unroll
      for (int j = 0; j < 8; ++j)
        s += qp[c * 8 + j] * (float)kv[j] * (float)dv[j];
    }
    s *= cmv;
    if (kpm[(size_t)bn * Nc + m]) s = 0.0f;
    probs[m * 8 + h] = siluf(s) * cos_cutoff(dist[(size_t)bn * Nc + m]);
  }
  __syncthreads();

  // ---- apn[m][e] = probs[h][m] * v[m][e]   (overwrite dk tile in place, f16, vectorized)
  for (int i = lane; i < 512; i += 32) {
    int ml = i >> 5, c8 = i & 31, m = mbase + ml, h = c8 >> 2;
    float pr = probs[m * 8 + h];
    f16x8 vv = *(const f16x8*)(vh + (size_t)(b * Nc + m) * Ec + c8 * 8);
    f16x8 o;
#pragma unroll
    for (int j = 0; j < 8; ++j) o[j] = (_Float16)(pr * (float)vv[j]);
    *(f16x8*)(apn_h + m * Ec + c8 * 8) = o;
  }
  __syncthreads();

  // ---- attn[b,n,e] = sum_m apn[m][e]  (vectorized row segments + LDS atomic reduce into probs[])
  float* attn_acc = probs;               // probs is dead now; reuse 256 floats
  attn_acc[tid] = 0.0f;
  __syncthreads();
  {
    int eb = tid & 31, mg = tid >> 5;    // 32 e-blocks of 8, 8 m-groups of 16
    float s0 = 0, s1 = 0, s2 = 0, s3 = 0, s4 = 0, s5 = 0, s6 = 0, s7 = 0;
    for (int mm = 0; mm < 16; ++mm) {
      f16x8 a = *(const f16x8*)(apn_h + (mg * 16 + mm) * Ec + eb * 8);
      s0 += (float)a[0]; s1 += (float)a[1]; s2 += (float)a[2]; s3 += (float)a[3];
      s4 += (float)a[4]; s5 += (float)a[5]; s6 += (float)a[6]; s7 += (float)a[7];
    }
    atomicAdd(&attn_acc[eb * 8 + 0], s0); atomicAdd(&attn_acc[eb * 8 + 1], s1);
    atomicAdd(&attn_acc[eb * 8 + 2], s2); atomicAdd(&attn_acc[eb * 8 + 3], s3);
    atomicAdd(&attn_acc[eb * 8 + 4], s4); atomicAdd(&attn_acc[eb * 8 + 5], s5);
    atomicAdd(&attn_acc[eb * 8 + 6], s6); atomicAdd(&attn_acc[eb * 8 + 7], s7);
  }
  __syncthreads();
  out[(size_t)bn * Ec + tid] = attn_acc[tid];

  // ---- du_msg = mask(apn @ Wdu + bdu);  du[s][e] += sum_m du_msg[m][e]*vec[m][s]
  {
    const _Float16* At = apn_h + mbase * Ec;
    for (int ct = 0; ct < 16; ++ct) {
      f32x8 acc = {};
#pragma unroll
      for (int kb = 0; kb < 8; ++kb)
        acc = wmma(load_a(At, l15, hi, kb * 32), load_b(WduT, ct, kb * 32, l15, hi), acc);
      int e = ct * 16 + l15;
      float bb = bdu[e];
      float p0 = 0.0f, p1 = 0.0f, p2 = 0.0f;
#pragma unroll
      for (int r = 0; r < 8; ++r) {
        int m = mbase + r + 8 * hi;
        float msg = acc[r] + bb;
        if (kpm[(size_t)bn * Nc + m]) msg = 0.0f;
        p0 += msg * vecs[m * 3 + 0];
        p1 += msg * vecs[m * 3 + 1];
        p2 += msg * vecs[m * 3 + 2];
      }
      atomicAdd(&du_acc[e], p0);
      atomicAdd(&du_acc[256 + e], p1);
      atomicAdd(&du_acc[512 + e], p2);
    }
  }
  __syncthreads();

  // ---- vec_rms_norm(du)
  {
    float d0 = du_acc[tid], d1 = du_acc[256 + tid], d2 = du_acc[512 + tid];
    float dd = sqrtf(d0 * d0 + d1 * d1 + d2 * d2);
    dd = fmaxf(dd, 1e-12f);
    atomicAdd(red, dd * dd);
  }
  __syncthreads();
  {
    float inv = 1.0f / sqrtf(red[0] * (1.0f / 256.0f));
    du_acc[tid] *= inv; du_acc[256 + tid] *= inv; du_acc[512 + tid] *= inv;
  }
  __syncthreads();

  // ---- wswt = du @ Wdih; center_ws = ws*cm -> cws; wt -> wtb
  for (int j = 0; j < 6; ++j) {
    int idx = tid + 256 * j;        // 0..1535
    int s = idx >> 9, o = idx & 511;
    const float* dn = du_acc + s * 256;
    float acc = 0.0f;
    for (int k = 0; k < 256; ++k) acc += dn[k] * Wdih[k * 512 + o];
    if (o < 256) cws[(size_t)bn * 768 + s * 256 + o] = acc * cmv;
    else         wtb[(size_t)bn * 768 + s * 256 + (o - 256)] = acc;
  }
}

// Pass 2: ipe[b,n,m,e] = silu(ea@Wea + bea) * sum_s cws[b,n,s,e]*wt[b,m,s,e]
__global__ __launch_bounds__(256) void pass2(
    const float* __restrict__ edge_attr, const _Float16* __restrict__ WeaT,
    const float* __restrict__ bea, const float* __restrict__ cws,
    const float* __restrict__ wtb, float* __restrict__ out) {
  extern __shared__ char smem[];
  float* cws_s = (float*)smem;                      // 3*256
  _Float16* ea_h = (_Float16*)(smem + 3072);        // 128*256 f16
  int tid = threadIdx.x, w = tid >> 5, lane = tid & 31, l15 = lane & 15, hi = lane >> 4;
  int bn = blockIdx.x, b = bn >> 7, mbase = w * 16;

  for (int i = tid; i < 768; i += 256) cws_s[i] = cws[(size_t)bn * 768 + i];
  stage16x256(edge_attr + (size_t)bn * Nc * Ec + (size_t)mbase * Ec, ea_h + mbase * Ec, lane);

  // prefetch this wave's 48KB wt tile (16 rows x 768 f32) -> global_prefetch_b8
  {
    const float* wtbase = wtb + (size_t)(b * Nc + mbase) * 768;
    for (int i = lane; i < 768; i += 32)           // one prefetch per 64B line
      __builtin_prefetch(wtbase + i * 16, 0, 3);
  }
  __syncthreads();

  const _Float16* At = ea_h + mbase * Ec;
  for (int ct = 0; ct < 16; ++ct) {
    f32x8 acc = {};
#pragma unroll
    for (int kb = 0; kb < 8; ++kb)
      acc = wmma(load_a(At, l15, hi, kb * 32), load_b(WeaT, ct, kb * 32, l15, hi), acc);
    int e = ct * 16 + l15;
    float be = bea[e];
    float c0 = cws_s[e], c1 = cws_s[256 + e], c2 = cws_s[512 + e];
#pragma unroll
    for (int r = 0; r < 8; ++r) {
      int m = mbase + r + 8 * hi;
      float g = siluf(acc[r] + be);
      const float* wtp = wtb + (size_t)(b * Nc + m) * 768 + e;
      float ip = g * (c0 * wtp[0] + c1 * wtp[256] + c2 * wtp[512]);
      out[(size_t)ATTN_SZ + ((size_t)bn * Nc + m) * Ec + e] = ip;
    }
  }
}

// ---------------------------------------------------------------- launch
extern "C" void kernel_launch(void* const* d_in, const int* in_sizes, int n_in,
                              void* d_out, int out_size, void* d_ws, size_t ws_size,
                              hipStream_t stream) {
  (void)in_sizes; (void)n_in; (void)out_size; (void)ws_size;
  const float* x     = (const float*)d_in[0];
  const float* vec   = (const float*)d_in[1];
  const float* dist  = (const float*)d_in[2];
  const float* ea    = (const float*)d_in[3];
  const unsigned char* kpm   = (const unsigned char*)d_in[4];
  const unsigned char* cmask = (const unsigned char*)d_in[5];
  const float* Wq = (const float*)d_in[6],  *bq  = (const float*)d_in[7];
  const float* Wk = (const float*)d_in[8],  *bk  = (const float*)d_in[9];
  const float* Wv = (const float*)d_in[10], *bv  = (const float*)d_in[11];
  const float* Wdk = (const float*)d_in[12], *bdk = (const float*)d_in[13];
  const float* Wdu = (const float*)d_in[14], *bdu = (const float*)d_in[15];
  const float* Wdih = (const float*)d_in[16];
  const float* Wea = (const float*)d_in[17], *bea = (const float*)d_in[18];
  float* out = (float*)d_out;

  char* base = (char*)d_ws; size_t off = 0;
  auto take = [&](size_t bytes) -> char* {
    char* r = base + off;
    off += (bytes + 255) & ~(size_t)255;
    return r;
  };
  _Float16* WqT  = (_Float16*)take(Ec * Ec * 2);
  _Float16* WkT  = (_Float16*)take(Ec * Ec * 2);
  _Float16* WvT  = (_Float16*)take(Ec * Ec * 2);
  _Float16* WdkT = (_Float16*)take(Ec * Ec * 2);
  _Float16* WduT = (_Float16*)take(Ec * Ec * 2);
  _Float16* WeaT = (_Float16*)take(Ec * Ec * 2);
  float*    q32  = (float*)   take((size_t)Bc * Nc * Ec * 4);
  _Float16* khb  = (_Float16*)take((size_t)Bc * Nc * Ec * 2);
  _Float16* vhb  = (_Float16*)take((size_t)Bc * Nc * Ec * 2);
  float*    cwsb = (float*)   take((size_t)Bc * Nc * 768 * 4);
  float*    wtbf = (float*)   take((size_t)Bc * Nc * 768 * 4);

  prep_wT<<<Ec, Ec, 0, stream>>>(Wq, WqT);
  prep_wT<<<Ec, Ec, 0, stream>>>(Wk, WkT);
  prep_wT<<<Ec, Ec, 0, stream>>>(Wv, WvT);
  prep_wT<<<Ec, Ec, 0, stream>>>(Wdk, WdkT);
  prep_wT<<<Ec, Ec, 0, stream>>>(Wdu, WduT);
  prep_wT<<<Ec, Ec, 0, stream>>>(Wea, WeaT);

  const int PROJ_SMEM = 64 + 2 * 16 * Ec * 2;                 // 16448 B
  proj_qkv<<<(Bc * Nc) / 16, 256, PROJ_SMEM, stream>>>(
      x, cmask, WqT, WkT, WvT, bq, bk, bv, q32, khb, vhb);

  const int P1_SMEM = 2440 * 4 + 2 * 128 * Ec * 2;            // 140832 B
  pass1<<<Bc * Nc, 256, P1_SMEM, stream>>>(
      q32, khb, vhb, ea, dist, kpm, cmask, vec,
      WdkT, bdk, WduT, bdu, Wdih, cwsb, wtbf, out);

  const int P2_SMEM = 768 * 4 + 128 * Ec * 2;                 // 68608 B
  pass2<<<Bc * Nc, 256, P2_SMEM, stream>>>(ea, WeaT, bea, cwsb, wtbf, out);
}
